// MultiHeadAttention_87282325389588
// MI455X (gfx1250) — matrix operand order repaired and checked by reading, not compiled, not run on hardware
//
#include <hip/hip_runtime.h>

// ---------------------------------------------------------------------------
// MHA forward for MI455X (gfx1250, wave32, WMMA).
// All heavy math on v_wmma_f32_16x16x32_bf16 (bf16 in, f32 accumulate).
// ---------------------------------------------------------------------------

typedef __attribute__((ext_vector_type(16))) __bf16 v16bf;
typedef __attribute__((ext_vector_type(8)))  __bf16 v8bf;
typedef __attribute__((ext_vector_type(8)))  float  v8f;

#define B_   2
#define T_   4096
#define C_   768
#define H_   12
#define DK_  64
#define M_   (B_ * T_)   // 8192
#define N3C_ (3 * C_)    // 2304

#define LOG2E 1.44269504088896340736f

__device__ __forceinline__ __bf16 f2bf(float f) {
  // round-to-nearest-even f32 -> bf16
  unsigned u = __builtin_bit_cast(unsigned, f);
  unsigned r = u + 0x7FFFu + ((u >> 16) & 1u);
  unsigned short h = (unsigned short)(r >> 16);
  return __builtin_bit_cast(__bf16, h);
}

__device__ __forceinline__ v16bf cat8(v8bf lo, v8bf hi) {
  return __builtin_shufflevector(lo, hi, 0,1,2,3,4,5,6,7,8,9,10,11,12,13,14,15);
}

__device__ __forceinline__ v8f wmma_bf16(v16bf a, v16bf b, v8f c) {
  return __builtin_amdgcn_wmma_f32_16x16x32_bf16(false, a, false, b, (short)0, c,
                                                 false, false);
}

// ---------------------------------------------------------------------------
// Stage 0: casts
// ---------------------------------------------------------------------------
__global__ __launch_bounds__(256) void cast_f32_bf16(const float* __restrict__ src,
                                                     __bf16* __restrict__ dst, int n) {
  int i = (blockIdx.x * 256 + threadIdx.x) * 4;
  if (i + 3 < n) {
    float4 v = *(const float4*)(src + i);
    dst[i + 0] = f2bf(v.x);
    dst[i + 1] = f2bf(v.y);
    dst[i + 2] = f2bf(v.z);
    dst[i + 3] = f2bf(v.w);
  }
}

// src is [K][N] f32 row-major, dst is [N][K] bf16 row-major (K-contiguous)
__global__ __launch_bounds__(256) void transpose_cast(const float* __restrict__ src,
                                                      __bf16* __restrict__ dst,
                                                      int K, int N) {
  int i = blockIdx.x * 256 + threadIdx.x;
  if (i < K * N) {
    int k = i / N, n = i - k * N;
    dst[(size_t)n * K + k] = f2bf(src[i]);
  }
}

// ---------------------------------------------------------------------------
// Stage 1: QKV GEMM.  X[8192][768] @ Wqkv -> scatter Q(*log2e/sqrt(Dk)),K,V
// bf16 into [B][H][T][Dk].  Block tile 64x256, BK=32, 8 waves,
// 2 WMMA cols per wave (A-fragment reuse).
// ---------------------------------------------------------------------------
__global__ __launch_bounds__(256) void qkv_gemm(const __bf16* __restrict__ X,
                                                const __bf16* __restrict__ WT,
                                                const float* __restrict__ bias,
                                                __bf16* __restrict__ Qb,
                                                __bf16* __restrict__ Kb,
                                                __bf16* __restrict__ Vb) {
  __shared__ __attribute__((aligned(64))) __bf16 sA[64][32];
  __shared__ __attribute__((aligned(64))) __bf16 sB[256][32];
  const int t = threadIdx.x, lane = t & 31, w = t >> 5;
  const int la = lane & 15, g = lane >> 4;
  const int m0 = blockIdx.y * 64, n0 = blockIdx.x * 256;

  v8f acc[2][4] = {};
  const int arow = t >> 2, aseg = (t & 3) * 8;

  for (int k0 = 0; k0 < C_; k0 += 32) {
    *(int4*)&sA[arow][aseg] = *(const int4*)(X + (size_t)(m0 + arow) * C_ + k0 + aseg);
#pragma unroll
    for (int it = 0; it < 4; ++it) {
      int idx = t + it * 256;
      int br = idx >> 2, bs = (idx & 3) * 8;
      *(int4*)&sB[br][bs] = *(const int4*)(WT + (size_t)(n0 + br) * C_ + k0 + bs);
    }
    __syncthreads();
    v16bf bfr[2];
#pragma unroll
    for (int jj = 0; jj < 2; ++jj)
      bfr[jj] = *(const v16bf*)&sB[(w * 2 + jj) * 16 + la][g * 16];
#pragma unroll
    for (int i4 = 0; i4 < 4; ++i4) {
      const __bf16* rp = &sA[i4 * 16 + la][0];
      v16bf af = cat8(*(const v8bf*)(rp + g * 16), *(const v8bf*)(rp + g * 16 + 8));
#pragma unroll
      for (int jj = 0; jj < 2; ++jj)
        acc[jj][i4] = wmma_bf16(af, bfr[jj], acc[jj][i4]);
    }
    __syncthreads();
  }

  // epilogue: bias; Q additionally scaled by log2e/sqrt(Dk) (softmax uses exp2)
#pragma unroll
  for (int jj = 0; jj < 2; ++jj) {
    const int n = n0 + (w * 2 + jj) * 16 + la;
    const float bv = bias[n];
    const int which = n / C_;
    const int c = n - which * C_;
    const int hh = c >> 6, d = c & 63;
    __bf16* dst = (which == 0) ? Qb : (which == 1 ? Kb : Vb);
    const float mul = (which == 0) ? (0.125f * LOG2E) : 1.0f;  // 1/sqrt(64)*log2e
#pragma unroll
    for (int i4 = 0; i4 < 4; ++i4) {
#pragma unroll
      for (int r = 0; r < 8; ++r) {
        int m = m0 + i4 * 16 + g * 8 + r;
        int bi = m >> 12;        // / 4096
        int tok = m & 4095;
        float val = (acc[jj][i4][r] + bv) * mul;
        dst[((size_t)(bi * H_ + hh) * T_ + tok) * DK_ + d] = f2bf(val);
      }
    }
  }
}

// ---------------------------------------------------------------------------
// Stage 2: flash attention.  Block = 8 waves x 16 q-rows = 128 q rows.
// 64-key steps, LDS double-buffered (one barrier/step), V stored transposed
// so P*V B-fragments are pairs of contiguous v8bf LDS loads.
// Computes S^T = K*Q^T so the WMMA f32 accumulator (lane = q col) IS the
// A-fragment layout needed for P*V after exp2 — no transpose round-trip.
// ---------------------------------------------------------------------------
__global__ __launch_bounds__(256) void attention(const __bf16* __restrict__ Qb,
                                                 const __bf16* __restrict__ Kb,
                                                 const __bf16* __restrict__ Vb,
                                                 __bf16* __restrict__ Ob) {
  __shared__ __attribute__((aligned(64))) __bf16 sK[2][64][72];   // [kk][d], +8 pad
  __shared__ __attribute__((aligned(64))) __bf16 sVT[2][64][80];  // [d][kk], +16 pad
  const int t = threadIdx.x, lane = t & 31, w = t >> 5;
  const int la = lane & 15, g = lane >> 4;
  const int bidx = blockIdx.z, h = blockIdx.y;
  const int bh = bidx * H_ + h;
  const int qw = blockIdx.x * 128 + w * 16;

  const __bf16* Qbh = Qb + (size_t)bh * T_ * DK_;
  const __bf16* Kbh = Kb + (size_t)bh * T_ * DK_;
  const __bf16* Vbh = Vb + (size_t)bh * T_ * DK_;

  // Q fragments (B-operand: lane = q col, K index = d) resident in VGPRs
  const __bf16* qrow = Qbh + (size_t)(qw + la) * DK_ + g * 16;
  v16bf bq0 = *(const v16bf*)qrow;         // d in [0,32)
  v16bf bq1 = *(const v16bf*)(qrow + 32);  // d in [32,64)

  v8f accO[4] = {};
  float mstate = -3.0e38f, lstate = 0.f;
  const int lkk = t >> 3, lseg = (t & 7) * 8;

  auto load_tile = [&](int kb, int pb) {
#pragma unroll
    for (int hh = 0; hh < 2; ++hh) {
      int row = lkk + hh * 32;  // key index within tile
      *(int4*)&sK[pb][row][lseg] =
          *(const int4*)(Kbh + (size_t)(kb + row) * DK_ + lseg);
      int4 vv = *(const int4*)(Vbh + (size_t)(kb + row) * DK_ + lseg);
      const __bf16* ve = (const __bf16*)&vv;
#pragma unroll
      for (int i = 0; i < 8; ++i) sVT[pb][lseg + i][row] = ve[i];  // transpose
    }
  };

  load_tile(0, 0);
  int pb = 0;
  for (int kb = 0; kb < T_; kb += 64, pb ^= 1) {
    __syncthreads();
    if (kb + 64 < T_) {
      load_tile(kb + 64, pb ^ 1);
      if (kb + 128 < T_) {
        __builtin_prefetch(Kbh + (size_t)(kb + 128 + lkk) * DK_ + lseg, 0, 0);
        __builtin_prefetch(Vbh + (size_t)(kb + 128 + lkk) * DK_ + lseg, 0, 0);
      }
    }

    // S^T tiles: 4 kk-subtiles x (2 chained d-chunks)
    v8f accT[4] = {};
#pragma unroll
    for (int s = 0; s < 4; ++s) {
      const __bf16* rp = &sK[pb][s * 16 + la][0];
      v16bf a0 = cat8(*(const v8bf*)(rp + g * 16),      *(const v8bf*)(rp + g * 16 + 8));
      v16bf a1 = cat8(*(const v8bf*)(rp + 32 + g * 16), *(const v8bf*)(rp + 40 + g * 16));
      accT[s] = wmma_bf16(a0, bq0, accT[s]);
      accT[s] = wmma_bf16(a1, bq1, accT[s]);
    }

    // online softmax in exp2 domain (Q was pre-scaled by log2e/sqrt(Dk));
    // lane holds q = la; the two 16-lane halves cover complementary kk sets
    float mloc = -3.0e38f;
#pragma unroll
    for (int s = 0; s < 4; ++s)
#pragma unroll
      for (int r = 0; r < 8; ++r) mloc = fmaxf(mloc, accT[s][r]);
    mloc = fmaxf(mloc, __shfl_xor(mloc, 16));
    float mnew = fmaxf(mstate, mloc);
    float corr = exp2f(mstate - mnew);

    v16bf pf0, pf1;
    float rsum = 0.f;
#pragma unroll
    for (int r = 0; r < 8; ++r) {
      float p0 = exp2f(accT[0][r] - mnew);   // kk =      8g + r
      float p1 = exp2f(accT[1][r] - mnew);   // kk = 16 + 8g + r
      float p2 = exp2f(accT[2][r] - mnew);   // kk = 32 + 8g + r
      float p3 = exp2f(accT[3][r] - mnew);   // kk = 48 + 8g + r
      rsum += (p0 + p1) + (p2 + p3);
      pf0[r] = f2bf(p0); pf0[8 + r] = f2bf(p1);
      pf1[r] = f2bf(p2); pf1[8 + r] = f2bf(p3);
    }
    rsum += __shfl_xor(rsum, 16);
    lstate = lstate * corr + rsum;
    mstate = mnew;

    // rescale O (O layout: lane row = 8g + r) then accumulate P*V
#pragma unroll
    for (int r = 0; r < 8; ++r) {
      float cr = __shfl(corr, g * 8 + r);
      accO[0][r] *= cr; accO[1][r] *= cr; accO[2][r] *= cr; accO[3][r] *= cr;
    }
#pragma unroll
    for (int tt = 0; tt < 4; ++tt) {
      const __bf16* vp = &sVT[pb][tt * 16 + la][0];
      v16bf b0 = cat8(*(const v8bf*)(vp + g * 8), *(const v8bf*)(vp + 16 + g * 8));        // kk in [0,32)
      v16bf b1 = cat8(*(const v8bf*)(vp + 32 + g * 8), *(const v8bf*)(vp + 48 + g * 8));   // kk in [32,64)
      accO[tt] = wmma_bf16(pf0, b0, accO[tt]);
      accO[tt] = wmma_bf16(pf1, b1, accO[tt]);
    }
  }

  // normalize and store bf16 to [B][T][C] (head-interleaved) for out-proj
  __bf16* obase = Ob + (size_t)bidx * T_ * C_ + (size_t)h * DK_;
#pragma unroll
  for (int r = 0; r < 8; ++r) {
    float li = 1.f / __shfl(lstate, g * 8 + r);
    int q = qw + g * 8 + r;
#pragma unroll
    for (int tt = 0; tt < 4; ++tt) {
      int d = tt * 16 + la;
      obase[(size_t)q * C_ + d] = f2bf(accO[tt][r] * li);
    }
  }
}

// ---------------------------------------------------------------------------
// Stage 3: output projection.  A[8192][768] bf16 @ WoutT -> f32 out (+bias)
// Block tile 64x256, 2 WMMA cols per wave.
// ---------------------------------------------------------------------------
__global__ __launch_bounds__(256) void out_gemm(const __bf16* __restrict__ A,
                                                const __bf16* __restrict__ WT,
                                                const float* __restrict__ bias,
                                                float* __restrict__ out) {
  __shared__ __attribute__((aligned(64))) __bf16 sA[64][32];
  __shared__ __attribute__((aligned(64))) __bf16 sB[256][32];
  const int t = threadIdx.x, lane = t & 31, w = t >> 5;
  const int la = lane & 15, g = lane >> 4;
  const int m0 = blockIdx.y * 64, n0 = blockIdx.x * 256;

  v8f acc[2][4] = {};
  const int arow = t >> 2, aseg = (t & 3) * 8;

  for (int k0 = 0; k0 < C_; k0 += 32) {
    *(int4*)&sA[arow][aseg] = *(const int4*)(A + (size_t)(m0 + arow) * C_ + k0 + aseg);
#pragma unroll
    for (int it = 0; it < 4; ++it) {
      int idx = t + it * 256;
      int br = idx >> 2, bs = (idx & 3) * 8;
      *(int4*)&sB[br][bs] = *(const int4*)(WT + (size_t)(n0 + br) * C_ + k0 + bs);
    }
    __syncthreads();
    v16bf bfr[2];
#pragma unroll
    for (int jj = 0; jj < 2; ++jj)
      bfr[jj] = *(const v16bf*)&sB[(w * 2 + jj) * 16 + la][g * 16];
#pragma unroll
    for (int i4 = 0; i4 < 4; ++i4) {
      const __bf16* rp = &sA[i4 * 16 + la][0];
      v16bf af = cat8(*(const v8bf*)(rp + g * 16), *(const v8bf*)(rp + g * 16 + 8));
#pragma unroll
      for (int jj = 0; jj < 2; ++jj)
        acc[jj][i4] = wmma_bf16(af, bfr[jj], acc[jj][i4]);
    }
    __syncthreads();
  }

#pragma unroll
  for (int jj = 0; jj < 2; ++jj) {
    const int n = n0 + (w * 2 + jj) * 16 + la;
    const float bv = bias[n];
#pragma unroll
    for (int i4 = 0; i4 < 4; ++i4) {
#pragma unroll
      for (int r = 0; r < 8; ++r) {
        int m = m0 + i4 * 16 + g * 8 + r;
        out[(size_t)m * C_ + n] = acc[jj][i4][r] + bv;
      }
    }
  }
}

// ---------------------------------------------------------------------------
// Host launcher
// ---------------------------------------------------------------------------
extern "C" void kernel_launch(void* const* d_in, const int* in_sizes, int n_in,
                              void* d_out, int out_size, void* d_ws, size_t ws_size,
                              hipStream_t stream) {
  const float* x     = (const float*)d_in[0];  // [2,4096,768]
  const float* Wqkv  = (const float*)d_in[1];  // [768,2304]
  const float* bqkv  = (const float*)d_in[2];  // [2304]
  const float* Wout  = (const float*)d_in[3];  // [768,768]
  const float* bout  = (const float*)d_in[4];  // [768]
  float* out = (float*)d_out;

  // workspace carve (all chunks 256B-aligned); total ~= 64.5 MB
  char* p = (char*)d_ws;
  __bf16* xb    = (__bf16*)p; p += (size_t)M_ * C_ * 2;       // 12.6 MB
  __bf16* wqkvT = (__bf16*)p; p += (size_t)N3C_ * C_ * 2;     //  3.5 MB
  __bf16* woutT = (__bf16*)p; p += (size_t)C_ * C_ * 2;       //  1.2 MB
  __bf16* Qb    = (__bf16*)p; p += (size_t)B_ * H_ * T_ * DK_ * 2;
  __bf16* Kb    = (__bf16*)p; p += (size_t)B_ * H_ * T_ * DK_ * 2;
  __bf16* Vb    = (__bf16*)p; p += (size_t)B_ * H_ * T_ * DK_ * 2;
  __bf16* Ob    = (__bf16*)p; p += (size_t)M_ * C_ * 2;
  (void)ws_size; (void)in_sizes; (void)n_in; (void)out_size;

  // stage 0: casts / transposes
  cast_f32_bf16<<<(M_ * C_) / (256 * 4), 256, 0, stream>>>(x, xb, M_ * C_);
  transpose_cast<<<(C_ * N3C_ + 255) / 256, 256, 0, stream>>>(Wqkv, wqkvT, C_, N3C_);
  transpose_cast<<<(C_ * C_ + 255) / 256, 256, 0, stream>>>(Wout, woutT, C_, C_);

  // stage 1: QKV projection + scatter (Q pre-scaled by log2e/sqrt(Dk))
  qkv_gemm<<<dim3(N3C_ / 256, M_ / 64), 256, 0, stream>>>(xb, wqkvT, bqkv, Qb, Kb, Vb);

  // stage 2: flash attention (exp2-domain softmax)
  attention<<<dim3(T_ / 128, H_, B_), 256, 0, stream>>>(Qb, Kb, Vb, Ob);

  // stage 3: output projection
  out_gemm<<<dim3(C_ / 128 / 2, M_ / 64), 256, 0, stream>>>(Ob, woutT, bout, out);
}
